// VectorQuantizer_26551487824163
// MI455X (gfx1250) — matrix operand order repaired and checked by reading, not compiled, not run on hardware
//
#include <hip/hip_runtime.h>
#include <hip/hip_bf16.h>

#define N_EMB 1024
#define CDIM  256
#define HW    4096          // 64*64
#define NPTS  131072        // 32*4096
#define ROWS_PER_WAVE  64   // 4 WMMA M-tiles held in registers
#define ROWS_PER_BLOCK 512  // 8 waves

typedef __attribute__((ext_vector_type(16))) __bf16       v16bf;
typedef __attribute__((ext_vector_type(8)))  float        v8f;
typedef __attribute__((ext_vector_type(8)))  unsigned int v8u;

__device__ __forceinline__ unsigned int f2bf(float f) {
    unsigned int u = __float_as_uint(f);
    unsigned int r = u + 0x7FFFu + ((u >> 16) & 1u);   // round-to-nearest-even
    return r >> 16;
}
__device__ __forceinline__ unsigned int packbf(float lo, float hi) {
    return f2bf(lo) | (f2bf(hi) << 16);                // even K in [15:0]
}

// CDNA5 async copy: global -> LDS, 16B per lane, tracked on ASYNCcnt
__device__ __forceinline__ void async_b128(unsigned int lds_off, const void* gaddr) {
    asm volatile("global_load_async_to_lds_b128 %0, %1, off"
                 :: "v"(lds_off), "v"(gaddr) : "memory");
}
#define WAIT_ASYNCCNT(n) asm volatile("s_wait_asynccnt %0" :: "n"(n) : "memory")

// ---------------- Kernel 1: codebook -> bf16, e_sq in f32 ----------------
__global__ void __launch_bounds__(256)
vq_prep(const float* __restrict__ emb,
        unsigned short* __restrict__ ebf,
        float* __restrict__ esq) {
    const int e = blockIdx.x;          // one codebook row per block
    const int c = threadIdx.x;         // 256 channels
    float v = emb[(size_t)e * CDIM + c];
    ebf[(size_t)e * CDIM + c] = (unsigned short)f2bf(v);
    float s = v * v;
    #pragma unroll
    for (int off = 16; off > 0; off >>= 1) s += __shfl_down(s, off, 32);
    __shared__ float part[8];
    const int wid = threadIdx.x >> 5, lane = threadIdx.x & 31;
    if (lane == 0) part[wid] = s;
    __syncthreads();
    if (threadIdx.x == 0) {
        float t = 0.f;
        #pragma unroll
        for (int i = 0; i < 8; ++i) t += part[i];
        esq[e] = t;
    }
}

// ---------------- Kernel 2: WMMA distance proxy + argmin ----------------
// proxy = e_sq - 2*(z . e); z_sq dropped (row-constant, argmin-invariant).
// 4 A-tiles per wave in VGPRs; B e-tiles double-buffered in LDS via
// global_load_async_to_lds_b128, shared by all 8 waves of the block.
__global__ void __launch_bounds__(256)
vq_argmin(const float* __restrict__ z,
          const unsigned short* __restrict__ ebf,
          const float* __restrict__ esq,
          int* __restrict__ idxout) {
    __shared__ __align__(16) unsigned short Bsm[2][16 * CDIM];   // 2 x 8KB

    const int tid  = threadIdx.x;
    const int lane = tid & 31;
    const int wave = tid >> 5;
    const int nblk = blockIdx.x * ROWS_PER_BLOCK;     // block's first point
    const int b    = nblk >> 12;                      // 4096 points per image
    const int hw0  = (nblk & 4095) + wave * ROWS_PER_WAVE;
    const int n0w  = nblk + wave * ROWS_PER_WAVE;     // wave's first point
    const float* zb = z + (size_t)b * CDIM * HW + hw0;

    const int h    = lane >> 4;                       // lane half -> K group
    const int mrow = lane & 15;                       // A: lane -> M row
    const int ncol = lane & 15;                       // B/D: lane -> N column

    // ---- build 4 x (16x256) bf16 A tiles once (contiguous k = 16h + s order) ----
    unsigned int afrag[4 * 64];
    #pragma unroll
    for (int t = 0; t < 4; ++t) {
        #pragma unroll
        for (int kb = 0; kb < 8; ++kb) {
            #pragma unroll
            for (int v = 0; v < 8; ++v) {
                const int koff = h * 16 + v * 2;
                const int c = kb * 32 + koff;
                const float f0 = zb[(size_t)c       * HW + t * 16 + mrow];
                const float f1 = zb[(size_t)(c + 1) * HW + t * 16 + mrow];
                afrag[t * 64 + kb * 8 + v] = packbf(f0, f1);
            }
        }
    }

    float bestd[4][8];
    int   beste[4][8];
    #pragma unroll
    for (int t = 0; t < 4; ++t)
        #pragma unroll
        for (int i = 0; i < 8; ++i) { bestd[t][i] = 3.4e38f; beste[t][i] = 0; }

    const unsigned int lds0 = (unsigned int)(uintptr_t)&Bsm[0][0];
    const unsigned int lds1 = (unsigned int)(uintptr_t)&Bsm[1][0];

    // prologue: tile 0 -> buffer 0 (each thread moves 2 x 16B of the 8KB tile)
    {
        const char* s = (const char*)ebf + tid * 16;
        async_b128(lds0 + tid * 16,        s);
        async_b128(lds0 + tid * 16 + 4096, s + 4096);
    }

    for (int et = 0; et < 64; ++et) {
        const int p = et & 1;
        // prefetch next e-tile into the other buffer (wraps harmlessly at end)
        {
            const int enext = ((et + 1) & 63) * 16;
            const char* s = (const char*)(ebf + (size_t)enext * CDIM) + tid * 16;
            const unsigned int d = (p ? lds0 : lds1) + tid * 16;
            async_b128(d,        s);
            async_b128(d + 4096, s + 4096);
        }
        WAIT_ASYNCCNT(2);     // our 2 older async ops (current buffer) done
        __syncthreads();      // every wave's slice of current buffer done

        v8f acc[4];
        #pragma unroll
        for (int t = 0; t < 4; ++t) {
            v8f zf = {0.f, 0.f, 0.f, 0.f, 0.f, 0.f, 0.f, 0.f};
            acc[t] = zf;
        }

        const unsigned short* bbase = &Bsm[p][(ncol << 8) + (h << 4)];
        #pragma unroll
        for (int kb = 0; kb < 8; ++kb) {
            const uint4* bp = (const uint4*)(bbase + kb * 32);
            uint4 b0 = bp[0], b1 = bp[1];
            v8u bu = {b0.x, b0.y, b0.z, b0.w, b1.x, b1.y, b1.z, b1.w};
            const v16bf bv = __builtin_bit_cast(v16bf, bu);
            #pragma unroll
            for (int t = 0; t < 4; ++t) {
                v8u au;
                #pragma unroll
                for (int v = 0; v < 8; ++v) au[v] = afrag[t * 64 + kb * 8 + v];
                acc[t] = __builtin_amdgcn_wmma_f32_16x16x32_bf16(
                    false, __builtin_bit_cast(v16bf, au),
                    false, bv,
                    (short)0, acc[t], false, false);
            }
        }

        const int   e0   = et * 16;
        const float es   = esq[e0 + ncol];
        const int   eidx = e0 + ncol;
        #pragma unroll
        for (int t = 0; t < 4; ++t) {
            #pragma unroll
            for (int i = 0; i < 8; ++i) {
                const float d = es - 2.0f * acc[t][i];   // row M=i+8h, col=ncol
                if (d < bestd[t][i]) { bestd[t][i] = d; beste[t][i] = eidx; }
            }
        }
        __syncthreads();      // all reads of buffer p done before its refill
    }

    // min-with-index across the 16 lanes holding the same output row
    #pragma unroll
    for (int t = 0; t < 4; ++t) {
        #pragma unroll
        for (int i = 0; i < 8; ++i) {
            #pragma unroll
            for (int s = 8; s >= 1; s >>= 1) {
                const float od = __shfl_xor(bestd[t][i], s, 32);
                const int   oi = __shfl_xor(beste[t][i], s, 32);
                if (od < bestd[t][i] ||
                    (od == bestd[t][i] && oi < beste[t][i])) {
                    bestd[t][i] = od; beste[t][i] = oi;
                }
            }
        }
    }
    if ((lane & 15) == 0) {
        #pragma unroll
        for (int t = 0; t < 4; ++t)
            #pragma unroll
            for (int i = 0; i < 8; ++i)
                idxout[n0w + t * 16 + i + 8 * h] = beste[t][i];
    }
}

// ---------------- Kernel 3: coalesced gather (NCHW output) ----------------
__global__ void __launch_bounds__(256)
vq_gather(const float* __restrict__ emb,
          const int* __restrict__ idx,
          float* __restrict__ out) {
    const size_t o = (size_t)blockIdx.x * 256 + threadIdx.x; // [b][c][hw]
    const int hw = (int)(o & 4095);
    const int bc = (int)(o >> 12);
    const int c  = bc & 255;
    const int b  = bc >> 8;
    const int code = idx[b * HW + hw];
    out[o] = emb[(size_t)code * CDIM + c];
}

extern "C" void kernel_launch(void* const* d_in, const int* in_sizes, int n_in,
                              void* d_out, int out_size, void* d_ws, size_t ws_size,
                              hipStream_t stream) {
    const float* z   = (const float*)d_in[0];   // [32,256,64,64]
    const float* emb = (const float*)d_in[1];   // [1024,256]
    float* out = (float*)d_out;                 // [32,256,64,64]

    char* ws = (char*)d_ws;
    int*            idx = (int*)ws;                                    // 512 KB
    unsigned short* ebf = (unsigned short*)(ws + (size_t)NPTS * 4);    // 512 KB
    float*          esq = (float*)(ws + (size_t)NPTS * 4
                                      + (size_t)N_EMB * CDIM * 2);     // 4 KB

    vq_prep  <<<N_EMB,                 256, 0, stream>>>(emb, ebf, esq);
    vq_argmin<<<NPTS / ROWS_PER_BLOCK, 256, 0, stream>>>(z, ebf, esq, idx);
    vq_gather<<<NPTS * (CDIM / 256),   256, 0, stream>>>(emb, idx, out);
}
